// Net_25082609009218
// MI455X (gfx1250) — matrix operand order repaired and checked by reading, not compiled, not run on hardware
//
#include <hip/hip_runtime.h>
#include <math.h>

typedef __attribute__((ext_vector_type(16))) _Float16 v16h;
typedef __attribute__((ext_vector_type(8)))  float    v8f;

// ---------------------------------------------------------------------------
// WMMA helpers (CDNA5 gfx1250, wave32). Layouts per CDNA5 ISA:
//  A (16x32 f16): lane 0-15 -> M=lane, VGPR j holds K pair (j<4? 2j : 16+2(j-4));
//                 lanes 16-31 add +8 to K.
//  B (32x16 f16): lane -> N=lane&15, VGPR j holds same K pair as A (kpatA); lanes>=16 add +8.
//  C (16x16 f32): VGPR r -> M = r + (lane>=16 ? 8 : 0), N = lane&15.
// ---------------------------------------------------------------------------
__device__ __forceinline__ v8f wmma16(v16h a, v16h b, v8f c) {
  return __builtin_amdgcn_wmma_f32_16x16x32_f16(false, a, false, b, (short)0, c,
                                                false, false);
}

__device__ __forceinline__ int kpatA(int j) { return (j < 4) ? 2 * j : 16 + 2 * (j - 4); }

// A fragment from a row-major f16 LDS tile (rows = M, leading dim ld).
__device__ __forceinline__ v16h ldsA(const _Float16* t, int ld, int k0, int lane) {
  v16h a; int m = lane & 15; int kb = (lane >> 4) * 8;
#pragma unroll
  for (int j = 0; j < 8; ++j) {
    int k = k0 + kpatA(j) + kb;
    a[2 * j]     = t[m * ld + k];
    a[2 * j + 1] = t[m * ld + k + 1];
  }
  return a;
}

// B fragment from a row-major f16 (K x N) matrix, leading dim ld.
__device__ __forceinline__ v16h ldB(const _Float16* t, int ld, int k0, int n0, int lane) {
  v16h b; int n = (lane & 15) + n0; int kb = (lane >> 4) * 8;
#pragma unroll
  for (int j = 0; j < 8; ++j) {
    int k = k0 + kpatA(j) + kb;
    b[2 * j]     = t[(size_t)k * ld + n];
    b[2 * j + 1] = t[(size_t)(k + 1) * ld + n];
  }
  return b;
}

// B fragment from f16 LDS, rows masked to k < Kv (zero-pad K dim to 32).
__device__ __forceinline__ v16h ldB_guard(const _Float16* t, int ld, int n0, int Kv, int lane) {
  v16h b; int n = (lane & 15) + n0; int kb = (lane >> 4) * 8;
#pragma unroll
  for (int j = 0; j < 8; ++j) {
    int k = kpatA(j) + kb;
    b[2 * j]     = (k < Kv)     ? t[k * ld + n]       : (_Float16)0.f;
    b[2 * j + 1] = (k + 1 < Kv) ? t[(k + 1) * ld + n] : (_Float16)0.f;
  }
  return b;
}

// Build A fragment of d = local[dim] - centers[dim][c], feature idx = dim*27+c,
// zero padded past 81. loc: [P][3] f32 LDS tile, cen: [3][27] f32 global.
__device__ __forceinline__ v16h buildA_d(const float* loc, const float* cen,
                                         int p0, int k0, int lane) {
  v16h a; int m = lane & 15; int kb = (lane >> 4) * 8; int p = p0 + m;
#pragma unroll
  for (int j = 0; j < 8; ++j) {
    int kbase = k0 + kpatA(j) + kb;
#pragma unroll
    for (int e = 0; e < 2; ++e) {
      int kk = kbase + e; float v = 0.f;
      if (kk < 81) { int dim = kk / 27; int cc = kk - dim * 27;
                     v = loc[p * 3 + dim] - cen[dim * 27 + cc]; }
      a[2 * j + e] = (_Float16)v;
    }
  }
  return a;
}

// A fragment = gathered neighbor features (f16 activations), rows = channels.
__device__ __forceinline__ v16h gatherA_feats(const _Float16* xb, const int* nidx,
                                              int Cin, int c0, int Kv, int lane) {
  v16h a; int m = lane & 15; int kb = (lane >> 4) * 8; int c = c0 + m;
#pragma unroll
  for (int j = 0; j < 8; ++j) {
    int k = kpatA(j) + kb;
    a[2 * j]     = (c < Cin && k < Kv)       ? xb[(size_t)nidx[k] * Cin + c]
                                             : (_Float16)0.f;
    a[2 * j + 1] = (c < Cin && (k + 1) < Kv) ? xb[(size_t)nidx[k + 1] * Cin + c]
                                             : (_Float16)0.f;
  }
  return a;
}

// C fragment -> (bias + relu) -> f16 row-major LDS tile; cols >= nvalid zeroed.
__device__ __forceinline__ void storeC_relu_f16(v8f c, _Float16* t, int ld, int n0,
                                                const float* bias, int nvalid, int lane) {
  int n = (lane & 15) + n0; int hi = lane >> 4;
  float bv = (n < nvalid) ? bias[n] : 0.f;
#pragma unroll
  for (int r = 0; r < 8; ++r) {
    float v = c[r] + bv; v = v > 0.f ? v : 0.f;
    t[(r + hi * 8) * ld + n] = (_Float16)((n < nvalid) ? v : 0.f);
  }
}

// ---------------------------------------------------------------------------
// KNN: one thread per (batch, query); insertion top-K (K <= 16).
// pts is input_pts (128 x 2048 x 3); each layer's point set is its prefix.
// ---------------------------------------------------------------------------
__global__ void knn_kernel(const float* __restrict__ pts, int N, int n_out, int K,
                           int* __restrict__ idx) {
  int t = blockIdx.x * blockDim.x + threadIdx.x;
  if (t >= 128 * n_out) return;
  int b = t / n_out, q = t - b * n_out;
  const float* pb = pts + (size_t)b * 2048 * 3;
  float qx = pb[q * 3 + 0], qy = pb[q * 3 + 1], qz = pb[q * 3 + 2];
  float bd[16]; int bi[16];
#pragma unroll
  for (int i = 0; i < 16; ++i) { bd[i] = 3.4e38f; bi[i] = 0; }
  for (int p = 0; p < N; ++p) {
    float dx = pb[p * 3 + 0] - qx, dy = pb[p * 3 + 1] - qy, dz = pb[p * 3 + 2] - qz;
    float d2 = dx * dx + dy * dy + dz * dz;
    if (d2 < bd[K - 1]) {
      int j = K - 1;
      while (j > 0 && bd[j - 1] > d2) { bd[j] = bd[j - 1]; bi[j] = bi[j - 1]; --j; }
      bd[j] = d2; bi[j] = p;
    }
  }
  for (int i = 0; i < K; ++i) idx[(size_t)t * K + i] = bi[i];
}

// ---------------------------------------------------------------------------
// Weight/activation prep: f32 -> padded row-major f16.
// ---------------------------------------------------------------------------
__global__ void pad16_kernel(const float* __restrict__ src, int R, int C,
                             _Float16* __restrict__ dst, int Rp, int Cp) {
  int t = blockIdx.x * blockDim.x + threadIdx.x;
  if (t >= Rp * Cp) return;
  int r = t / Cp, c = t - r * Cp;
  dst[t] = (_Float16)((r < R && c < C) ? src[r * C + c] : 0.f);
}

// Conv weight (Cin*27, cout) -> [nct][448][cout] f16, 448-row slices per
// 16-channel tile (slice row r = c_local*27 + m for r < 432, else zero).
__global__ void prep_convw_kernel(const float* __restrict__ w, int Cin, int cout,
                                  _Float16* __restrict__ dst, int nct) {
  int t = blockIdx.x * blockDim.x + threadIdx.x;
  int total = nct * 448 * cout;
  if (t >= total) return;
  int o = t % cout; int r = (t / cout) % 448; int ct = t / (cout * 448);
  float v = 0.f;
  if (r < 432) {
    int cl = r / 27, m = r - cl * 27, c = ct * 16 + cl;
    if (c < Cin) v = w[((size_t)c * 27 + m) * cout + o];
  }
  dst[t] = (_Float16)v;
}

// ---------------------------------------------------------------------------
// Fused point-conv layer: one wave handles 16 queries of one batch.
//  gather local coords -> per-query radius norm -> MLP (WMMA f16, K pads
//  96/64/32) -> per-query f = featsT @ h (WMMA) -> out = F @ Wconv (WMMA).
// ---------------------------------------------------------------------------
__global__ void __launch_bounds__(32)
ptconv_kernel(const _Float16* __restrict__ x, const float* __restrict__ pts,
              const int* __restrict__ idx, const float* __restrict__ cen,
              const _Float16* __restrict__ l1f, const _Float16* __restrict__ l2f,
              const _Float16* __restrict__ l3f, const float* __restrict__ l1b,
              const float* __restrict__ l2b, const float* __restrict__ l3b,
              const _Float16* __restrict__ wconv, const float* __restrict__ bconv,
              float* __restrict__ xout, int N, int Cin, int n_out, int K, int cout,
              int nct, int qtiles) {
  __shared__ float    s_loc[16 * 16 * 3];   // local coords per (q,k)
  __shared__ int      s_idx[16 * 16];       // neighbor indices, stride 16
  __shared__ _Float16 s_h[16 * 16 * 32];    // MLP output h, [p][32]
  __shared__ _Float16 s_t1[16 * 64];        // h1 staging
  __shared__ _Float16 s_t2[16 * 32];        // h2 staging
  __shared__ _Float16 s_F[16 * 448];        // per-ctile F slice, [q][448]

  int lane = threadIdx.x & 31;
  int bb = blockIdx.x / qtiles;
  int q0 = (blockIdx.x - bb * qtiles) * 16;
  const _Float16* xb = x + (size_t)bb * N * Cin;

  // ---- gather neighbor indices and raw local coords -----------------------
  for (int t = lane; t < 16 * K; t += 32) {
    int ql = t / K, k = t - ql * K; int q = q0 + ql;
    int id = 0; float lx = 0.f, ly = 0.f, lz = 0.f;
    if (q < n_out) {
      id = idx[((size_t)bb * n_out + q) * K + k];
      const float* pq = pts + ((size_t)bb * 2048 + q) * 3;
      const float* pp = pts + ((size_t)bb * 2048 + id) * 3;
      lx = pp[0] - pq[0]; ly = pp[1] - pq[1]; lz = pp[2] - pq[2];
    }
    s_idx[ql * 16 + k] = id;
    s_loc[t * 3 + 0] = lx; s_loc[t * 3 + 1] = ly; s_loc[t * 3 + 2] = lz;
  }
  __syncthreads();

  // ---- per-query radius normalization (lanes 0..15, one query each) -------
  if (lane < 16) {
    float mx = 0.f;
    for (int k = 0; k < K; ++k) {
      float dx = s_loc[(lane * K + k) * 3 + 0];
      float dy = s_loc[(lane * K + k) * 3 + 1];
      float dz = s_loc[(lane * K + k) * 3 + 2];
      float ss = dx * dx + dy * dy + dz * dz;
      mx = ss > mx ? ss : mx;
    }
    float mi = sqrtf(mx); if (!(mi > 0.f)) mi = 1.f;
    float inv = 1.f / mi;
    for (int k = 0; k < K; ++k)
      for (int d = 0; d < 3; ++d) s_loc[(lane * K + k) * 3 + d] *= inv;
  }
  __syncthreads();

  // ---- MLP over all 16*K neighbor points, 16 rows per WMMA tile -----------
#pragma unroll 1
  for (int ch = 0; ch < K; ++ch) {
    int p0 = ch * 16;
    v16h a0 = buildA_d(s_loc, cen, p0, 0,  lane);
    v16h a1 = buildA_d(s_loc, cen, p0, 32, lane);
    v16h a2 = buildA_d(s_loc, cen, p0, 64, lane);
    // layer 1: 96 -> 64 (54 valid)
#pragma unroll
    for (int nt = 0; nt < 4; ++nt) {
      v8f c = {};
      c = wmma16(a0, ldB(l1f, 64, 0,  nt * 16, lane), c);
      c = wmma16(a1, ldB(l1f, 64, 32, nt * 16, lane), c);
      c = wmma16(a2, ldB(l1f, 64, 64, nt * 16, lane), c);
      storeC_relu_f16(c, s_t1, 64, nt * 16, l1b, 54, lane);
    }
    // layer 2: 64 -> 32 (27 valid)
    v16h h10 = ldsA(s_t1, 64, 0, lane);
    v16h h11 = ldsA(s_t1, 64, 32, lane);
#pragma unroll
    for (int nt = 0; nt < 2; ++nt) {
      v8f c = {};
      c = wmma16(h10, ldB(l2f, 32, 0,  nt * 16, lane), c);
      c = wmma16(h11, ldB(l2f, 32, 32, nt * 16, lane), c);
      storeC_relu_f16(c, s_t2, 32, nt * 16, l2b, 27, lane);
    }
    // layer 3: 32 -> 32 (27 valid), write into h tile for this chunk
    v16h h2a = ldsA(s_t2, 32, 0, lane);
#pragma unroll
    for (int nt = 0; nt < 2; ++nt) {
      v8f c = wmma16(h2a, ldB(l3f, 32, 0, nt * 16, lane), (v8f){});
      storeC_relu_f16(c, s_h + p0 * 32, 32, nt * 16, l3b, 27, lane);
    }
  }

  // ---- zero pad rows 432..447 of F slices ---------------------------------
  for (int t = lane; t < 16 * 16; t += 32) {
    int q = t / 16; s_F[q * 448 + 432 + (t & 15)] = (_Float16)0.f;
  }
  __syncthreads();

  // ---- f = featsT @ h per query, then out += F @ Wconv --------------------
  int nout_t = cout >> 4;
  v8f zero = {};
  v8f acc[8];
#pragma unroll
  for (int i = 0; i < 8; ++i) acc[i] = zero;

#pragma unroll 1
  for (int ct = 0; ct < nct; ++ct) {
    const _Float16* wc = wconv + (size_t)ct * 448 * cout;
    __builtin_prefetch((const void*)wc, 0, 0);  // global_prefetch_b8
#pragma unroll 1
    for (int ql = 0; ql < 16; ++ql) {
      const _Float16* hq = s_h + ql * K * 32;
      v16h bh0 = ldB_guard(hq, 32, 0,  K, lane);
      v16h bh1 = ldB_guard(hq, 32, 16, K, lane);
      v16h af  = gatherA_feats(xb, s_idx + ql * 16, Cin, ct * 16, K, lane);
      v8f f0 = wmma16(af, bh0, zero);
      v8f f1 = wmma16(af, bh1, zero);
      int n = lane & 15, hi = lane >> 4;
#pragma unroll
      for (int r = 0; r < 8; ++r)
        s_F[ql * 448 + (r + hi * 8) * 27 + n] = (_Float16)f0[r];
      if (n + 16 < 27) {
#pragma unroll
        for (int r = 0; r < 8; ++r)
          s_F[ql * 448 + (r + hi * 8) * 27 + 16 + n] = (_Float16)f1[r];
      }
    }
    __syncthreads();
#pragma unroll
    for (int nt = 0; nt < 8; ++nt) {
      if (nt < nout_t) {
        v8f a = acc[nt];
#pragma unroll 1
        for (int kc = 0; kc < 14; ++kc) {
          v16h aF = ldsA(s_F, 448, kc * 32, lane);
          v16h bW = ldB(wc, cout, kc * 32, nt * 16, lane);
          a = wmma16(aF, bW, a);
        }
        acc[nt] = a;
      }
    }
    __syncthreads();
  }

  // ---- finalize: /K + bias, store pre-BN output (f32) ---------------------
  float invK = 1.f / (float)K;
  int n = lane & 15, hi = lane >> 4;
#pragma unroll
  for (int nt = 0; nt < 8; ++nt) {
    if (nt < nout_t) {
      int o = nt * 16 + n;
      float bv = bconv[o];
#pragma unroll
      for (int r = 0; r < 8; ++r) {
        int q = r + hi * 8;
        if (q0 + q < n_out)
          xout[((size_t)bb * n_out + q0 + q) * cout + o] = acc[nt][r] * invK + bv;
      }
    }
  }
}

// ---------------------------------------------------------------------------
// BatchNorm: per-channel block reduce over pre-BN f32; apply writes f16
// activations for the next layer's WMMA gathers.
// ---------------------------------------------------------------------------
__global__ void bn_stats_kernel(const float* __restrict__ x, int Nrows, int cout,
                                float* __restrict__ stats) {
  __shared__ float ssum[256], ssq[256];
  int co = blockIdx.x;
  float s = 0.f, s2 = 0.f;
  for (int i = threadIdx.x; i < Nrows; i += blockDim.x) {
    float v = x[(size_t)i * cout + co];
    s += v; s2 += v * v;
  }
  ssum[threadIdx.x] = s; ssq[threadIdx.x] = s2;
  __syncthreads();
  for (int st = 128; st > 0; st >>= 1) {
    if (threadIdx.x < st) {
      ssum[threadIdx.x] += ssum[threadIdx.x + st];
      ssq[threadIdx.x]  += ssq[threadIdx.x + st];
    }
    __syncthreads();
  }
  if (threadIdx.x == 0) {
    float m = ssum[0] / (float)Nrows;
    float var = ssq[0] / (float)Nrows - m * m;
    stats[co] = m;
    stats[cout + co] = rsqrtf(var + 1e-5f);
  }
}

__global__ void bn_apply_kernel(const float* __restrict__ xin,
                                _Float16* __restrict__ xh, int Nrows, int cout,
                                const float* __restrict__ stats,
                                const float* __restrict__ g,
                                const float* __restrict__ b) {
  int t = blockIdx.x * blockDim.x + threadIdx.x;
  if (t >= Nrows * cout) return;
  int co = t % cout;
  float v = (xin[t] - stats[co]) * stats[cout + co] * g[co] + b[co];
  xh[t] = (_Float16)(v > 0.f ? v : 0.f);
}

// ---------------------------------------------------------------------------
// Final FC: (128 x 1024) @ (1024 x 48pad) + bias. A-tile staged through LDS
// with CDNA5 async global->LDS copies (ASYNCcnt), both operands f16.
// ---------------------------------------------------------------------------
__global__ void __launch_bounds__(32)
fc_kernel(const _Float16* __restrict__ x, const _Float16* __restrict__ w48,
          const float* __restrict__ bias, float* __restrict__ out) {
  __shared__ _Float16 s_x[16 * 32];
  int lane = threadIdx.x & 31;
  int m0 = blockIdx.x * 16;
  int row = lane >> 2, seg = lane & 3;  // 8 rows x 4x16B segments per async op
  v8f acc0 = {}, acc1 = {}, acc2 = {};
#pragma unroll 1
  for (int kc = 0; kc < 1024; kc += 32) {
    // async-copy 16x32 f16 A-tile into LDS: 2 x b128 per lane (ASYNCcnt path)
    unsigned la0 = (unsigned)(size_t)&s_x[row * 32 + seg * 8];
    unsigned la1 = (unsigned)(size_t)&s_x[(row + 8) * 32 + seg * 8];
    unsigned long long ga0 =
        (unsigned long long)(x + (size_t)(m0 + row) * 1024 + kc + seg * 8);
    unsigned long long ga1 =
        (unsigned long long)(x + (size_t)(m0 + row + 8) * 1024 + kc + seg * 8);
    asm volatile("global_load_async_to_lds_b128 %0, %1, off"
                 :: "v"(la0), "v"(ga0) : "memory");
    asm volatile("global_load_async_to_lds_b128 %0, %1, off"
                 :: "v"(la1), "v"(ga1) : "memory");
    asm volatile("s_wait_asynccnt 0" ::: "memory");
    __syncthreads();
    v16h a = ldsA(s_x, 32, 0, lane);
    acc0 = wmma16(a, ldB(w48, 48, kc, 0,  lane), acc0);
    acc1 = wmma16(a, ldB(w48, 48, kc, 16, lane), acc1);
    acc2 = wmma16(a, ldB(w48, 48, kc, 32, lane), acc2);
    __syncthreads();
  }
  int n = lane & 15, hi = lane >> 4;
#pragma unroll
  for (int nt = 0; nt < 3; ++nt) {
    v8f c = nt == 0 ? acc0 : (nt == 1 ? acc1 : acc2);
    int o = nt * 16 + n;
    if (o < 40) {
      float bv = bias[o];
#pragma unroll
      for (int r = 0; r < 8; ++r)
        out[(size_t)(m0 + r + hi * 8) * 40 + o] = c[r] + bv;
    }
  }
}

// ---------------------------------------------------------------------------
// Host driver
// ---------------------------------------------------------------------------
extern "C" void kernel_launch(void* const* d_in, const int* in_sizes, int n_in,
                              void* d_out, int out_size, void* d_ws, size_t ws_size,
                              hipStream_t stream) {
  (void)in_sizes; (void)n_in; (void)out_size; (void)ws_size;
  static const int Ks[5]    = {16, 16, 8, 8, 4};
  static const int NOs[5]   = {1024, 256, 64, 16, 8};
  static const int CINs[5]  = {3, 64, 64, 64, 128};
  static const int COUTs[5] = {64, 64, 64, 128, 128};
  static const int NPs[5]   = {2048, 1024, 256, 64, 16};

  const float* x0   = (const float*)d_in[0];
  const float* pts  = (const float*)d_in[1];
  const float* fc_w = (const float*)d_in[57];
  const float* fc_b = (const float*)d_in[58];
  auto LP = [&](int l, int item) { return (const float*)d_in[2 + l * 11 + item]; };
  // items: 0=w 1=b 2=centers 3=l1w 4=l1b 5=l2w 6=l2b 7=l3w 8=l3b 9=bng 10=bnb

  char* ws = (char*)d_ws; size_t off = 0;
  auto carve = [&](size_t bytes) -> void* {
    off = (off + 255) & ~(size_t)255;
    void* p = ws + off; off += bytes; return p;
  };
  int*      idxb  = (int*)carve((size_t)128 * 1024 * 16 * sizeof(int));
  float*    xF    = (float*)carve((size_t)128 * 1024 * 64 * sizeof(float));  // pre-BN
  _Float16* xh0   = (_Float16*)carve((size_t)128 * 2048 * 3 * sizeof(_Float16));
  _Float16* xhA   = (_Float16*)carve((size_t)128 * 1024 * 64 * sizeof(_Float16));
  _Float16* xhB   = (_Float16*)carve((size_t)128 * 1024 * 64 * sizeof(_Float16));
  float*    stats = (float*)carve(256 * sizeof(float));
  _Float16* fcw48 = (_Float16*)carve((size_t)1024 * 48 * sizeof(_Float16));
  _Float16 *l1f[5], *l2f[5], *l3f[5], *cw[5];
  for (int l = 0; l < 5; ++l) {
    l1f[l] = (_Float16*)carve(96 * 64 * sizeof(_Float16));
    l2f[l] = (_Float16*)carve(64 * 32 * sizeof(_Float16));
    l3f[l] = (_Float16*)carve(32 * 32 * sizeof(_Float16));
    int nct = (CINs[l] + 15) / 16;
    cw[l] = (_Float16*)carve((size_t)nct * 448 * COUTs[l] * sizeof(_Float16));
  }

  // layer-0 input activations -> f16
  {
    int total = 128 * 2048 * 3;
    pad16_kernel<<<(total + 255) / 256, 256, 0, stream>>>(x0, 1, total, xh0, 1, total);
  }
  // fc weight -> f16 padded 1024 x 48
  pad16_kernel<<<(1024 * 48 + 255) / 256, 256, 0, stream>>>(fc_w, 1024, 40, fcw48, 1024, 48);

  const _Float16* xh_cur = xh0;
  _Float16* xhbufs[2] = {xhA, xhB};
  for (int l = 0; l < 5; ++l) {
    int K = Ks[l], n_out = NOs[l], Cin = CINs[l], cout = COUTs[l], N = NPs[l];
    int nct = (Cin + 15) / 16;
    int qtiles = (n_out + 15) / 16;
    _Float16* xh_next = xhbufs[l & 1];

    // weight prep (f32 -> padded f16)
    pad16_kernel<<<(96 * 64 + 255) / 256, 256, 0, stream>>>(LP(l, 3), 81, 54, l1f[l], 96, 64);
    pad16_kernel<<<(64 * 32 + 255) / 256, 256, 0, stream>>>(LP(l, 5), 54, 27, l2f[l], 64, 32);
    pad16_kernel<<<(32 * 32 + 255) / 256, 256, 0, stream>>>(LP(l, 7), 27, 27, l3f[l], 32, 32);
    {
      int total = nct * 448 * cout;
      prep_convw_kernel<<<(total + 255) / 256, 256, 0, stream>>>(LP(l, 0), Cin, cout, cw[l], nct);
    }

    knn_kernel<<<(128 * n_out + 255) / 256, 256, 0, stream>>>(pts, N, n_out, K, idxb);

    ptconv_kernel<<<128 * qtiles, 32, 0, stream>>>(
        xh_cur, pts, idxb, LP(l, 2), l1f[l], l2f[l], l3f[l], LP(l, 4), LP(l, 6),
        LP(l, 8), cw[l], LP(l, 1), xF, N, Cin, n_out, K, cout, nct, qtiles);

    bn_stats_kernel<<<cout, 256, 0, stream>>>(xF, 128 * n_out, cout, stats);
    {
      int total = 128 * n_out * cout;
      bn_apply_kernel<<<(total + 255) / 256, 256, 0, stream>>>(
          xF, xh_next, 128 * n_out, cout, stats, LP(l, 9), LP(l, 10));
    }
    xh_cur = xh_next;
  }

  // Final FC: x (128 x 1024, f16) @ fc_w (1024 x 48pad, f16) + fc_b
  fc_kernel<<<8, 32, 0, stream>>>(xh_cur, fcw48, fc_b, (float*)d_out);
}
